// VectorQuantizer_57681410785796
// MI455X (gfx1250) — matrix operand order repaired and checked by reading, not compiled, not run on hardware
//
#include <hip/hip_runtime.h>
#include <hip/hip_bf16.h>

// ---------------------------------------------------------------------------
// VQ-VAE eval forward for MI455X (gfx1250, wave32, WMMA + async global->LDS).
//
// argmin_k ||x - e_k||^2  ==  argmax_k ( x . e_k - 0.5*||e_k||^2 )
// Search in bf16 WMMA (f32 accumulate); losses exact f32 on gathered rows.
//
// N = 16*2048 = 32768 rows, D = 256, K = 8192 codes.
// Pipeline: prep (f32->bf16 codebook + half-norms, once) -> fused
// matmul/argmax with double-buffered async B-tile staging -> gather+MSE.
// ---------------------------------------------------------------------------

#define VQ_N      32768
#define VQ_D      256
#define VQ_K      8192
#define MTILE     128        // rows per workgroup (8 waves x 16 rows)
#define NTILE     16         // codes per iteration (one WMMA N)
#define KCHUNKS   8          // D / 32 per-WMMA reduction steps
#define BSTRIDE   264        // LDS row stride in bf16 (132 dwords -> conflict-free)

typedef __attribute__((ext_vector_type(16))) __bf16          v16bf;
typedef __attribute__((ext_vector_type(8)))  float           v8f;
typedef __attribute__((ext_vector_type(16))) unsigned short  v16us;
typedef __attribute__((ext_vector_type(8)))  unsigned short  v8us;

__device__ __forceinline__ unsigned short f2bf_rne(float f) {
    unsigned u = __float_as_uint(f);
    u += 0x7FFFu + ((u >> 16) & 1u);          // round-to-nearest-even
    return (unsigned short)(u >> 16);
}

__device__ __forceinline__ v8us cvt8(float4 a, float4 b) {
    v8us u;
    u[0] = f2bf_rne(a.x); u[1] = f2bf_rne(a.y); u[2] = f2bf_rne(a.z); u[3] = f2bf_rne(a.w);
    u[4] = f2bf_rne(b.x); u[5] = f2bf_rne(b.y); u[6] = f2bf_rne(b.z); u[7] = f2bf_rne(b.w);
    return u;
}

__device__ __forceinline__ v16bf pack16(v8us lo, v8us hi) {
    v16us u;
#pragma unroll
    for (int i = 0; i < 8; ++i) { u[i] = lo[i]; u[i + 8] = hi[i]; }
    return __builtin_bit_cast(v16bf, u);
}

// ---------------------------------------------------------------------------
// Kernel 1: per codebook row -> bf16 copy + ehn[k] = 0.5*||e_k||^2.
// One wave per row; lane holds 8 consecutive f32.
// ---------------------------------------------------------------------------
__global__ __launch_bounds__(256) void vq_prep_kernel(const float* __restrict__ emb,
                                                      unsigned short* __restrict__ embh,
                                                      float* __restrict__ ehn) {
    const int lane = threadIdx.x & 31;
    const int wave = threadIdx.x >> 5;
    const int row  = blockIdx.x * 8 + wave;
    const float4* r4 = (const float4*)(emb + (size_t)row * VQ_D) + lane * 2;
    float4 a = r4[0], b = r4[1];
    *(v8us*)(embh + (size_t)row * VQ_D + lane * 8) = cvt8(a, b);   // bf16 codebook
    float s = a.x * a.x + a.y * a.y + a.z * a.z + a.w * a.w +
              b.x * b.x + b.y * b.y + b.z * b.z + b.w * b.w;
#pragma unroll
    for (int off = 16; off >= 1; off >>= 1) s += __shfl_xor(s, off, 32);
    if (lane == 0) ehn[row] = 0.5f * s;
}

// ---------------------------------------------------------------------------
// Async-stage one 16-code bf16 tile (16 x 256 bf16, padded stride) into LDS.
// 256 threads x 32B each: two global_load_async_to_lds_b128 per thread
// (the instruction offset applies to both the LDS and the global address,
//  so the offset:16 form covers the second 16B on both sides). ASYNCcnt += 2/wave.
// ---------------------------------------------------------------------------
__device__ __forceinline__ void stage_tile_async(const unsigned short* __restrict__ embh,
                                                 unsigned short* lbuf, int kt, int tid) {
    const int row = tid >> 4;            // 0..15 codebook row within tile
    const int c   = (tid & 15) * 16;     // bf16 column
    const unsigned short* g = embh + (size_t)(kt + row) * VQ_D + c;
    unsigned lds = (unsigned)(uintptr_t)(lbuf + row * BSTRIDE + c);  // low 32b = LDS offset
    asm volatile("global_load_async_to_lds_b128 %0, %1, off"
                 :: "v"(lds), "v"(g) : "memory");
    asm volatile("global_load_async_to_lds_b128 %0, %1, off offset:16"
                 :: "v"(lds), "v"(g) : "memory");
}

// ---------------------------------------------------------------------------
// Kernel 2: fused (x @ E^T) + running argmax via bf16 WMMA.
//   grid = N/MTILE = 256 blocks, block = 256 threads (8 waves).
//   Wave w owns rows [blk*128 + 16w, +16); loops over K in 16-code tiles with
//   double-buffered async staging.
//
// WMMA register layouts (wave32, CDNA5 ISA):
//   A (16x32 bf16): lane l holds row l%16; elems 0..7 -> k = 8*(l/16)+0..7,
//                   elems 8..15 -> k = 8*(l/16)+16..23  (per 32-wide chunk)
//   B (32x16 bf16): lane l holds column l%16; elem i -> same k as A elem i
//   C/D (16x16 f32): VGPR r, lane l -> (M = r + 8*(l/16), N = l%16)
// ---------------------------------------------------------------------------
__global__ __launch_bounds__(256) void vq_argmin_kernel(const float* __restrict__ x,
                                                        const unsigned short* __restrict__ embh,
                                                        const float* __restrict__ ehn,
                                                        int* __restrict__ idx_out) {
    __shared__ __align__(16) unsigned short lds_b[2][NTILE * BSTRIDE];
    __shared__ float lds_ehn[VQ_K];

    const int tid  = threadIdx.x;
    const int lane = tid & 31;
    const int wave = tid >> 5;
    const int half = lane >> 4;
    const int l16  = lane & 15;

    const int row0  = blockIdx.x * MTILE + wave * 16;
    const int myrow = row0 + l16;

    // preload all half-norms into LDS (read per tile via ds_load, no loadcnt stall)
#pragma unroll 4
    for (int i = tid; i < VQ_K; i += 256) lds_ehn[i] = ehn[i];

    // ---- A fragments: this wave's 16 rows, f32 -> bf16, kept in registers ----
    v16bf afrag[KCHUNKS];
    {
        const float4* xr = (const float4*)(x + (size_t)myrow * VQ_D);
#pragma unroll
        for (int c = 0; c < KCHUNKS; ++c) {
            const int k0 = c * 32 + half * 8;
            float4 f0 = xr[k0 / 4],        f1 = xr[k0 / 4 + 1];
            float4 f2 = xr[(k0 + 16) / 4], f3 = xr[(k0 + 16) / 4 + 1];
            afrag[c] = pack16(cvt8(f0, f1), cvt8(f2, f3));
        }
    }

    float best[8];
    int   bidx[8];
#pragma unroll
    for (int r = 0; r < 8; ++r) { best[r] = -3.0e38f; bidx[r] = 0; }

    const int NT = VQ_K / NTILE;                 // 512 tiles
    stage_tile_async(embh, lds_b[0], 0, tid);    // prologue: tile 0 -> buf 0
    __syncthreads();                             // lds_ehn ready (and aligns waves)

    for (int t = 0; t < NT; ++t) {
        const int kt  = t * NTILE;
        const int ktn = (t + 1 < NT) ? kt + NTILE : 0;        // dummy re-stage at end
        stage_tile_async(embh, lds_b[(t + 1) & 1], ktn, tid); // overlap with compute
        asm volatile("s_wait_asynccnt 2" ::: "memory");       // in-order: cur tile done
        __syncthreads();                                      // visible to all waves

        const unsigned short* bbuf = lds_b[t & 1];
        const float h = lds_ehn[kt + l16];                    // 0.5*||e||^2 for my code

        v8f acc = {};
#pragma unroll
        for (int c = 0; c < KCHUNKS; ++c) {
            const unsigned short* src = &bbuf[l16 * BSTRIDE + c * 32 + half * 8];
            v8us b0 = *(const v8us*)(src);
            v8us b1 = *(const v8us*)(src + 16);
            acc = __builtin_amdgcn_wmma_f32_16x16x32_bf16(
                false, afrag[c], false, pack16(b0, b1), (short)0, acc, false, false);
        }

        const int code = kt + l16;
#pragma unroll
        for (int r = 0; r < 8; ++r) {
            float s = acc[r] - h;                 // dot - 0.5||e||^2
            if (s > best[r]) { best[r] = s; bidx[r] = code; }  // strict > keeps lowest idx
        }
        __syncthreads();    // all waves done reading buf[t&1] before it is re-staged
    }

    // per-row argmax across the 16 code lanes of each half-wave
#pragma unroll
    for (int r = 0; r < 8; ++r) {
        float b  = best[r];
        int   bi = bidx[r];
#pragma unroll
        for (int off = 8; off >= 1; off >>= 1) {
            float ob = __shfl_xor(b, off, 16);
            int   oi = __shfl_xor(bi, off, 16);
            if (ob > b || (ob == b && oi < bi)) { b = ob; bi = oi; }
        }
        if (l16 == 0) idx_out[row0 + r + half * 8] = bi;
    }
}

// ---------------------------------------------------------------------------
// Kernel 3: gather quantized rows (exact f32), write outputs, accumulate MSE.
// ---------------------------------------------------------------------------
__global__ __launch_bounds__(256) void vq_gather_loss_kernel(const float* __restrict__ x,
                                                             const float* __restrict__ emb,
                                                             const int* __restrict__ idx,
                                                             float* __restrict__ qout,
                                                             float* __restrict__ idxf,
                                                             float* __restrict__ accum) {
    const int tid  = threadIdx.x;
    const int row  = blockIdx.x * 4 + (tid >> 6);
    const int j    = tid & 63;
    const int lane = tid & 31;

    const int k = idx[row];
    float4 q  = ((const float4*)(emb + (size_t)k   * VQ_D))[j];
    float4 xv = ((const float4*)(x   + (size_t)row * VQ_D))[j];
    ((float4*)(qout + (size_t)row * VQ_D))[j] = q;   // straight-through fwd == quantized

    float dx = q.x - xv.x, dy = q.y - xv.y, dz = q.z - xv.z, dw = q.w - xv.w;
    float s  = dx * dx + dy * dy + dz * dz + dw * dw;
#pragma unroll
    for (int off = 16; off >= 1; off >>= 1) s += __shfl_xor(s, off, 32);
    if (lane == 0) atomicAdd(accum, s);
    if (j == 0)    idxf[row] = (float)k;
}

// ---------------------------------------------------------------------------
// Kernel 4: scalars. e/q latent losses are numerically equal in eval mode.
// ---------------------------------------------------------------------------
__global__ void vq_finalize_kernel(const float* __restrict__ accum, float* __restrict__ out3) {
    const float mse = accum[0] * (1.0f / ((float)VQ_N * (float)VQ_D));
    out3[0] = 1.25f * mse;   // vq_loss
    out3[1] = 0.25f * mse;   // commitment_loss
    out3[2] = mse;           // q_latent_loss
}

// ---------------------------------------------------------------------------
extern "C" void kernel_launch(void* const* d_in, const int* in_sizes, int n_in,
                              void* d_out, int out_size, void* d_ws, size_t ws_size,
                              hipStream_t stream) {
    (void)in_sizes; (void)n_in; (void)out_size; (void)ws_size;

    const float* x   = (const float*)d_in[0];   // (16,2048,256) f32
    const float* emb = (const float*)d_in[1];   // (8192,256)    f32
    float* out = (float*)d_out;

    // d_out layout (flat, return order): quantized_st | indices(f32) | vq | commit | q_latent
    float* out_q    = out;
    float* out_idx  = out + (size_t)VQ_N * VQ_D;            // 8388608
    float* out_scal = out_idx + VQ_N;                       // 8421376

    // workspace: bf16 codebook (4MB, 16B-aligned first) | idx[N] | ehn[K] | accum
    unsigned short* ws_embh = (unsigned short*)d_ws;
    int*   ws_idx   = (int*)((char*)d_ws + (size_t)VQ_K * VQ_D * sizeof(unsigned short));
    float* ws_ehn   = (float*)((char*)ws_idx + (size_t)VQ_N * sizeof(int));
    float* ws_accum = (float*)((char*)ws_ehn + (size_t)VQ_K * sizeof(float));

    hipMemsetAsync(ws_accum, 0, sizeof(float), stream);

    vq_prep_kernel<<<VQ_K / 8, 256, 0, stream>>>(emb, ws_embh, ws_ehn);
    vq_argmin_kernel<<<VQ_N / MTILE, 256, 0, stream>>>(x, ws_embh, ws_ehn, ws_idx);
    vq_gather_loss_kernel<<<VQ_N / 4, 256, 0, stream>>>(x, emb, ws_idx, out_q, out_idx, ws_accum);
    vq_finalize_kernel<<<1, 1, 0, stream>>>(ws_accum, out_scal);
}
